// Transformer_74105365725481
// MI455X (gfx1250) — matrix operand order repaired and checked by reading, not compiled, not run on hardware
//
#include <hip/hip_runtime.h>

// ---------------------------------------------------------------------------
// CDNA5 (gfx1250) transformer forward. wave32, WMMA 16x16x32 bf16, f32 accum,
// double-buffered K=64 LDS stages fed by GLOBAL_LOAD_ASYNC_TO_LDS_B128.
// ---------------------------------------------------------------------------

typedef __bf16 bf16;
typedef __attribute__((ext_vector_type(16))) __bf16 v16bf;
typedef __attribute__((ext_vector_type(8)))  float  v8f;

#define KSTEP     64
#define LDS_PAD   8               // 64+8 bf16 = 144B row stride (16B aligned)
#define TILE_ROWE (KSTEP + LDS_PAD)
#define TILE_ROWB (TILE_ROWE * 2) // bytes per LDS tile row
#define TILE_BYTES (128 * TILE_ROWB)

// async global->LDS copy, 16B per lane; ASYNCcnt-tracked (CDNA5)
#define ASYNC_CP_B128(ldsaddr, goff, base)                                   \
  asm volatile("global_load_async_to_lds_b128 %0, %1, %2"                    \
               :: "v"(ldsaddr), "v"(goff), "s"(base) : "memory")
#define WAIT_ASYNC0() asm volatile("s_wait_asynccnt 0x0" ::: "memory")

// ---------------------------------------------------------------------------
// Tiled GEMM: C[M,N] (+bias, +relu) = A[M,K](bf16) @ B(bf16)
//   TRANS_B=0 : B is [K,N] row-major      TRANS_B=1 : B is [N,K] (A @ B^T)
//   CSKIP     : causal, skip block tiles entirely above the diagonal (S GEMM)
//   CKLIM     : causal, clamp K loop to m0+128 (O = P@V, P zero past diag)
// Block tile 128x128, 8 waves; each wave owns 64x32 = 4x2 WMMA tiles.
// K staged 64 at a time (two 32-wide sub-phases -> 16 WMMAs per barrier).
// LDS: As[stage][128][64] row-major, Bs[stage][128][64] as Bs[n][k] so both
// fragment loads are contiguous b128 reads matching ISA bf16 layouts:
//   A 16x32: lanes 0-15 row M=l, K {0..7,16..23}; lanes 16-31 K {8..15,24..31}
//   B 32x16: lanes 0-15 col N=l, K {0..7,16..23}; lanes 16-31 K {8..15,24..31}
//   C/D f32: vgpr r -> M = r + 8*(lane>=16), N = lane&15
// ---------------------------------------------------------------------------
template<bool TRANS_B, bool BIAS, bool RELU, bool OUT_F32, bool OUT_BF16,
         bool CSKIP, bool CKLIM>
__global__ __launch_bounds__(256)
void gemm_wmma_bf16(const bf16* __restrict__ A, const bf16* __restrict__ B,
                    const float* __restrict__ bias,
                    float* __restrict__ Cf, bf16* __restrict__ Cb,
                    int M, int N, int K,
                    long long sA, long long sB, long long sC) {
  const int m0 = blockIdx.y * 128;
  const int n0 = blockIdx.x * 128;
  if (CSKIP && n0 > m0) return;               // tile fully masked by causality

  __shared__ __align__(16) bf16 As[2][128][TILE_ROWE];
  __shared__ __align__(16) bf16 Bs[2][128][TILE_ROWE];   // Bs[n][k]

  const int tid  = threadIdx.x;
  const int lane = tid & 31;
  const int wave = tid >> 5;
  const int wm   = (wave >> 2) * 64;          // wave row offset (2 waves in M)
  const int wn   = (wave & 3) * 32;           // wave col offset (4 waves in N)

  A += (size_t)blockIdx.z * sA;
  B += (size_t)blockIdx.z * sB;
  const int lda = K;
  const int ldb = TRANS_B ? K : N;

  const int l15 = lane & 15;
  const int hi  = lane >> 4;                  // 0 | 1
  const int kA  = hi * 8;                     // A-fragment K base
  const int kB  = hi * 8;                     // B-fragment K base

  const unsigned asBase = (unsigned)(uintptr_t)(&As[0][0][0]);
  const unsigned bsBase = (unsigned)(uintptr_t)(&Bs[0][0][0]);

  const int Kend = CKLIM ? ((m0 + 128 < K) ? m0 + 128 : K) : K;

  union bchunk { float4 f; bf16 h[8]; } breg[4];

  // issue stage loads for K-slice [k0, k0+64) into LDS stage `st`
  auto prefetch = [&](int k0, int st) {
    #pragma unroll
    for (int i = 0; i < 4; ++i) {             // A tile 128x64: async copy
      int c = tid + i * 256;                  // 1024 chunks of 8 bf16
      int r = c >> 3, c8 = (c & 7) * 8;
      unsigned lds  = asBase + (unsigned)st * TILE_BYTES + r * TILE_ROWB + c8 * 2;
      unsigned goff = (unsigned)(((m0 + r) * lda + k0 + c8) * 2);
      ASYNC_CP_B128(lds, goff, A);
    }
    if (TRANS_B) {                            // B^T tile: contiguous in K
      #pragma unroll
      for (int i = 0; i < 4; ++i) {
        int c = tid + i * 256;
        int n = c >> 3, c8 = (c & 7) * 8;
        unsigned lds  = bsBase + (unsigned)st * TILE_BYTES + n * TILE_ROWB + c8 * 2;
        unsigned goff = (unsigned)(((n0 + n) * ldb + k0 + c8) * 2);
        ASYNC_CP_B128(lds, goff, B);
      }
    } else {                                  // B tile needs transpose: regs
      #pragma unroll
      for (int i = 0; i < 4; ++i) {
        int c = tid + i * 256;                // 64 k-rows x 16 chunks
        int kk = c >> 4, nn = (c & 15) * 8;
        breg[i].f = *(const float4*)&B[(size_t)(k0 + kk) * ldb + n0 + nn];
      }
    }
  };
  // scatter-store the register-held B chunks into LDS stage `st`
  auto commitB = [&](int st) {
    if (!TRANS_B) {
      #pragma unroll
      for (int i = 0; i < 4; ++i) {
        int c = tid + i * 256;
        int kk = c >> 4, nn = (c & 15) * 8;
        #pragma unroll
        for (int j = 0; j < 8; ++j) Bs[st][nn + j][kk] = breg[i].h[j];
      }
    }
  };

  v8f acc[4][2] = {};

  prefetch(0, 0);
  commitB(0);
  WAIT_ASYNC0();
  __syncthreads();

  int cur = 0;
  for (int k0 = 0; k0 < Kend; k0 += KSTEP) {
    const bool hasNext = (k0 + KSTEP) < Kend;
    if (hasNext) prefetch(k0 + KSTEP, cur ^ 1);  // DMA overlaps WMMAs below

    #pragma unroll
    for (int kh = 0; kh < 2; ++kh) {          // two 32-wide sub-phases
      const int kb = kh * 32;
      union frag { v16bf v; float4 f[2]; } fa[4], fb[2];
      #pragma unroll
      for (int tm = 0; tm < 4; ++tm) {
        int r = wm + tm * 16 + l15;
        fa[tm].f[0] = *(const float4*)&As[cur][r][kb + kA];
        fa[tm].f[1] = *(const float4*)&As[cur][r][kb + kA + 16];
      }
      #pragma unroll
      for (int tn = 0; tn < 2; ++tn) {
        int c = wn + tn * 16 + l15;
        fb[tn].f[0] = *(const float4*)&Bs[cur][c][kb + kB];
        fb[tn].f[1] = *(const float4*)&Bs[cur][c][kb + kB + 16];
      }
      #pragma unroll
      for (int tm = 0; tm < 4; ++tm)
        #pragma unroll
        for (int tn = 0; tn < 2; ++tn)
          acc[tm][tn] = __builtin_amdgcn_wmma_f32_16x16x32_bf16(
              false, fa[tm].v, false, fb[tn].v, (short)0, acc[tm][tn],
              false, false);
    }

    if (hasNext) {
      commitB(cur ^ 1);                       // latency hidden under WMMAs
      WAIT_ASYNC0();
      __syncthreads();
    }
    cur ^= 1;
  }

  // ---- epilogue
  if (OUT_F32)  Cf += (size_t)blockIdx.z * sC;
  if (OUT_BF16) Cb += (size_t)blockIdx.z * sC;
  #pragma unroll
  for (int tm = 0; tm < 4; ++tm)
    #pragma unroll
    for (int tn = 0; tn < 2; ++tn) {
      const int colg = n0 + wn + tn * 16 + l15;
      float bv = BIAS ? bias[colg] : 0.0f;
      #pragma unroll
      for (int r = 0; r < 8; ++r) {
        const int rowg = m0 + wm + tm * 16 + hi * 8 + r;
        float v = acc[tm][tn][r];
        if (BIAS) v += bv;
        if (RELU) v = fmaxf(v, 0.0f);
        const size_t idx = (size_t)rowg * N + colg;
        if (OUT_F32)  Cf[idx] = v;
        if (OUT_BF16) Cb[idx] = (bf16)v;
      }
    }
}

// ---------------------------------------------------------------------------
// h = emb[x] + pos  (fp32 + bf16 copies)
// ---------------------------------------------------------------------------
__global__ __launch_bounds__(256)
void embed_kernel(const int* __restrict__ x, const float* __restrict__ emb,
                  const float* __restrict__ pos,
                  float* __restrict__ Hf, bf16* __restrict__ Hb,
                  int T, int D) {
  const int row = blockIdx.x;
  const int t   = row % T;
  const int idx = x[row];
  const float* e = emb + (size_t)idx * D;
  const float* p = pos + (size_t)t * D;
  const size_t base = (size_t)row * D;
  for (int d = threadIdx.x; d < D; d += blockDim.x) {
    float v = e[d] + p[d];
    Hf[base + d] = v;
    Hb[base + d] = (bf16)v;
  }
}

__global__ __launch_bounds__(256)
void cvt_f32_bf16(const float* __restrict__ in, bf16* __restrict__ out,
                  long long n) {
  long long i = (long long)blockIdx.x * blockDim.x + threadIdx.x;
  const long long stride = (long long)gridDim.x * blockDim.x;
  for (; i < n; i += stride) out[i] = (bf16)in[i];
}

__global__ __launch_bounds__(256)
void fill_bf16_zero(bf16* __restrict__ p, long long n) {
  long long i = (long long)blockIdx.x * blockDim.x + threadIdx.x;
  const long long stride = (long long)gridDim.x * blockDim.x;
  for (; i < n; i += stride) p[i] = (bf16)0.0f;
}

// ---------------------------------------------------------------------------
// causal softmax (scale folded in); writes only the valid prefix of each row
// (masked suffix of P is pre-zeroed once and never touched again)
// ---------------------------------------------------------------------------
__global__ __launch_bounds__(256)
void softmax_causal(const float* __restrict__ S, bf16* __restrict__ P,
                    int T, float scale) {
  const int t = blockIdx.x, b = blockIdx.y, tid = threadIdx.x;
  const size_t off = ((size_t)b * T + t) * (size_t)T;
  const float* srow = S + off;
  bf16* prow = P + off;
  const int valid = t + 1;

  __shared__ float red[256];
  float lmax = -3.0e38f;
  for (int i = tid; i < valid; i += 256) lmax = fmaxf(lmax, srow[i] * scale);
  red[tid] = lmax; __syncthreads();
  for (int s = 128; s > 0; s >>= 1) {
    if (tid < s) red[tid] = fmaxf(red[tid], red[tid + s]);
    __syncthreads();
  }
  const float m = red[0]; __syncthreads();

  float lsum = 0.0f;
  for (int i = tid; i < valid; i += 256) lsum += __expf(srow[i] * scale - m);
  red[tid] = lsum; __syncthreads();
  for (int s = 128; s > 0; s >>= 1) {
    if (tid < s) red[tid] += red[tid + s];
    __syncthreads();
  }
  const float inv = 1.0f / red[0];

  for (int i = tid; i < valid; i += 256)
    prow[i] = (bf16)(__expf(srow[i] * scale - m) * inv);
}

// ---------------------------------------------------------------------------
// h = LN(R + Z; gamma, beta), biased variance, eps=1e-5; fp32 + bf16 outputs
// ---------------------------------------------------------------------------
__global__ __launch_bounds__(256)
void add_layernorm(const float* __restrict__ R, const float* __restrict__ Z,
                   const float* __restrict__ g, const float* __restrict__ bta,
                   float* __restrict__ Hf, bf16* __restrict__ Hb, int D) {
  const int row = blockIdx.x, tid = threadIdx.x;
  const size_t base = (size_t)row * D + (size_t)tid * 4;
  float4 r4 = *(const float4*)&R[base];
  float4 z4 = *(const float4*)&Z[base];
  float x[4] = {r4.x + z4.x, r4.y + z4.y, r4.z + z4.z, r4.w + z4.w};

  __shared__ float red[256];
  red[tid] = x[0] + x[1] + x[2] + x[3]; __syncthreads();
  for (int s = 128; s > 0; s >>= 1) {
    if (tid < s) red[tid] += red[tid + s];
    __syncthreads();
  }
  const float mean = red[0] / (float)D; __syncthreads();

  red[tid] = x[0]*x[0] + x[1]*x[1] + x[2]*x[2] + x[3]*x[3]; __syncthreads();
  for (int s = 128; s > 0; s >>= 1) {
    if (tid < s) red[tid] += red[tid + s];
    __syncthreads();
  }
  const float var = red[0] / (float)D - mean * mean;
  const float inv = rsqrtf(var + 1e-5f);

  #pragma unroll
  for (int j = 0; j < 4; ++j) {
    const int d = tid * 4 + j;
    const float y = g[d] * (x[j] - mean) * inv + bta[d];
    Hf[base + j] = y;
    Hb[base + j] = (bf16)y;
  }
}

// ---------------------------------------------------------------------------
// host-side orchestration
// ---------------------------------------------------------------------------
extern "C" void kernel_launch(void* const* d_in, const int* in_sizes, int n_in,
                              void* d_out, int out_size, void* d_ws,
                              size_t ws_size, hipStream_t stream) {
  (void)in_sizes; (void)n_in; (void)out_size; (void)ws_size;
  const int Dm = 1024, Hm = 4096, Tt = 2048, Bb = 4, NL = 2;
  const int MT = Bb * Tt;                       // 8192 token rows

  const int*   x     = (const int*)  d_in[0];
  const float* emb   = (const float*)d_in[1];
  const float* pos   = (const float*)d_in[2];
  const float* wq    = (const float*)d_in[3];
  const float* wk    = (const float*)d_in[4];
  const float* wv    = (const float*)d_in[5];
  const float* wo    = (const float*)d_in[6];
  const float* wup   = (const float*)d_in[7];
  const float* bup   = (const float*)d_in[8];
  const float* wdown = (const float*)d_in[9];
  const float* bdown = (const float*)d_in[10];
  const float* ga    = (const float*)d_in[11];
  const float* ba    = (const float*)d_in[12];
  const float* gm    = (const float*)d_in[13];
  const float* bm    = (const float*)d_in[14];
  float* out = (float*)d_out;

  char* cur = (char*)d_ws;
  auto alloc = [&](size_t bytes) -> void* {
    void* p = cur;
    cur += (bytes + 255) & ~(size_t)255;
    return p;
  };
  float* h   = (float*)alloc((size_t)MT * Dm * 4);
  bf16*  hb  = (bf16*) alloc((size_t)MT * Dm * 2);
  bf16*  qb  = (bf16*) alloc((size_t)MT * Dm * 2);
  bf16*  kb  = (bf16*) alloc((size_t)MT * Dm * 2);
  bf16*  vb  = (bf16*) alloc((size_t)MT * Dm * 2);
  bf16*  ob  = (bf16*) alloc((size_t)MT * Dm * 2);
  float* S   = (float*)alloc((size_t)Bb * Tt * Tt * 4);
  bf16*  P   = (bf16*) alloc((size_t)Bb * Tt * Tt * 2);
  float* z   = (float*)alloc((size_t)MT * Dm * 4);   // reused for z and m
  bf16*  up  = (bf16*) alloc((size_t)MT * Hm * 2);
  bf16*  wqb = (bf16*) alloc((size_t)Dm * Dm * 2);
  bf16*  wkb = (bf16*) alloc((size_t)Dm * Dm * 2);
  bf16*  wvb = (bf16*) alloc((size_t)Dm * Dm * 2);
  bf16*  wob = (bf16*) alloc((size_t)Dm * Dm * 2);
  bf16*  wupb   = (bf16*)alloc((size_t)Dm * Hm * 2);
  bf16*  wdownb = (bf16*)alloc((size_t)Hm * Dm * 2);

  embed_kernel<<<MT, 256, 0, stream>>>(x, emb, pos, h, hb, Tt, Dm);
  fill_bf16_zero<<<4096, 256, 0, stream>>>(P, (long long)Bb * Tt * Tt);

  const dim3 gProj(Dm / 128, MT / 128, 1);            // [8192,1024,K]
  const dim3 gUp  (Hm / 128, MT / 128, 1);            // [8192,4096,1024]
  const dim3 gS   (Tt / 128, Tt / 128, Bb);           // per-batch [2048,2048,1024]
  const dim3 gO   (Dm / 128, Tt / 128, Bb);           // per-batch [2048,1024,2048]
  const long long aTD = (long long)Tt * Dm;
  const long long aTT = (long long)Tt * Tt;

  for (int l = 0; l < NL; ++l) {
    const size_t wOff  = (size_t)l * Dm * Dm;
    const size_t upOff = (size_t)l * Dm * Hm;
    cvt_f32_bf16<<<2048, 256, 0, stream>>>(wq + wOff, wqb, (long long)Dm * Dm);
    cvt_f32_bf16<<<2048, 256, 0, stream>>>(wk + wOff, wkb, (long long)Dm * Dm);
    cvt_f32_bf16<<<2048, 256, 0, stream>>>(wv + wOff, wvb, (long long)Dm * Dm);
    cvt_f32_bf16<<<2048, 256, 0, stream>>>(wo + wOff, wob, (long long)Dm * Dm);
    cvt_f32_bf16<<<4096, 256, 0, stream>>>(wup + upOff, wupb, (long long)Dm * Hm);
    cvt_f32_bf16<<<4096, 256, 0, stream>>>(wdown + upOff, wdownb, (long long)Hm * Dm);

    // Q,K,V projections (bf16 outputs for the attention WMMAs)
    gemm_wmma_bf16<false, false, false, false, true, false, false>
        <<<gProj, 256, 0, stream>>>(hb, wqb, nullptr, nullptr, qb,
                                    MT, Dm, Dm, 0, 0, 0);
    gemm_wmma_bf16<false, false, false, false, true, false, false>
        <<<gProj, 256, 0, stream>>>(hb, wkb, nullptr, nullptr, kb,
                                    MT, Dm, Dm, 0, 0, 0);
    gemm_wmma_bf16<false, false, false, false, true, false, false>
        <<<gProj, 256, 0, stream>>>(hb, wvb, nullptr, nullptr, vb,
                                    MT, Dm, Dm, 0, 0, 0);

    // S = Q @ K^T  (batched; upper-diagonal tiles skipped)
    gemm_wmma_bf16<true, false, false, true, false, true, false>
        <<<gS, 256, 0, stream>>>(qb, kb, nullptr, S, nullptr,
                                 Tt, Tt, Dm, aTD, aTD, aTT);

    // causal softmax with 1/sqrt(D) scale -> bf16 P (valid prefix only)
    softmax_causal<<<dim3(Tt, Bb), 256, 0, stream>>>(S, P, Tt, 0.03125f);

    // O = P @ V  (batched; K clamped to diagonal extent)
    gemm_wmma_bf16<false, false, false, false, true, false, true>
        <<<gO, 256, 0, stream>>>(P, vb, nullptr, nullptr, ob,
                                 Tt, Dm, Tt, aTT, aTD, aTD);

    // z = O @ Wo  (fp32 for the residual)
    gemm_wmma_bf16<false, false, false, true, false, false, false>
        <<<gProj, 256, 0, stream>>>(ob, wob, nullptr, z, nullptr,
                                    MT, Dm, Dm, 0, 0, 0);

    // h = LN(h + z)
    add_layernorm<<<MT, 256, 0, stream>>>(h, z,
        ga + (size_t)l * Dm, ba + (size_t)l * Dm, h, hb, Dm);

    // up = relu(h @ Wup + bup)  (bf16)
    gemm_wmma_bf16<false, true, true, false, true, false, false>
        <<<gUp, 256, 0, stream>>>(hb, wupb, bup + (size_t)l * Hm,
                                  nullptr, up, MT, Hm, Dm, 0, 0, 0);

    // m = up @ Wdown + bdown  (fp32, reuses z slab)
    gemm_wmma_bf16<false, true, false, true, false, false, false>
        <<<gProj, 256, 0, stream>>>(up, wdownb, bdown + (size_t)l * Dm,
                                    z, nullptr, MT, Dm, Hm, 0, 0, 0);

    // h = LN(h + m); final layer writes straight to d_out
    float* dst = (l == NL - 1) ? out : h;
    add_layernorm<<<MT, 256, 0, stream>>>(h, z,
        gm + (size_t)l * Dm, bm + (size_t)l * Dm, dst, hb, Dm);
  }
}
